// Attention_8323646620066
// MI455X (gfx1250) — matrix operand order repaired and checked by reading, not compiled, not run on hardware
//
#include <hip/hip_runtime.h>

// Problem constants (match reference)
#define NB    64      // batch
#define S_LEN 1024    // S_in == S_out
#define HDIM  512     // H

typedef __attribute__((ext_vector_type(16))) __bf16        v16bf;
typedef __attribute__((ext_vector_type(2)))  __bf16        v2bf;
typedef __attribute__((ext_vector_type(8)))  float         v8f;
typedef __attribute__((ext_vector_type(8)))  unsigned int  v8u;

// LDS tile geometry: rows x 32 bf16 == rows x 16 dwords, padded to 20 dwords
// (80B rows: 16B-aligned for b128 async writes; row*20+d conflict-free mod 64)
#define LDS_PITCH 20

// ---------------------------------------------------------------- utilities
// f32 pair -> packed bf16 dword via native converts (v_cvt_pk_bf16_f32 class)
__device__ __forceinline__ unsigned bf16pair(float lo, float hi) {
  v2bf p;
  p.x = (__bf16)lo;
  p.y = (__bf16)hi;
  return __builtin_bit_cast(unsigned, p);
}
__device__ __forceinline__ unsigned short bf16one(float f) {
  return __builtin_bit_cast(unsigned short, (__bf16)f);
}

// Register prefetch of a ROWSx32 f32 tile (row-major), stored as bf16 pairs.
template <int ROWS>
struct RowPF {
  static constexpr int N = (ROWS * 16) / 256;
  float2 v[N];
  __device__ __forceinline__ void load(const float* __restrict__ src, long stride, int tid) {
#pragma unroll
    for (int i = 0; i < N; ++i) {
      int p = tid + i * 256;
      int row = p >> 4, d = p & 15;
      v[i] = *(const float2*)(src + (long)row * stride + 2 * d);
    }
  }
  __device__ __forceinline__ void store(unsigned* lds, int tid) {
#pragma unroll
    for (int i = 0; i < N; ++i) {
      int p = tid + i * 256;
      int row = p >> 4, d = p & 15;
      lds[row * LDS_PITCH + d] = bf16pair(v[i].x, v[i].y);
    }
  }
};

// Register prefetch of B[k][n] (k in [0,32), n in [0,COLS)) from row-major
// src [32 x stride], K-pair packed so B fragments read contiguous dwords.
template <int COLS>
struct KnPF {
  static constexpr int N = (COLS * 16) / 256;
  float lo[N], hi[N];
  __device__ __forceinline__ void load(const float* __restrict__ src, long stride, int tid) {
#pragma unroll
    for (int i = 0; i < N; ++i) {
      int p = tid + i * 256;
      int n = p % COLS, d = p / COLS;
      lo[i] = src[(long)(2 * d) * stride + n];
      hi[i] = src[(long)(2 * d + 1) * stride + n];
    }
  }
  __device__ __forceinline__ void store(unsigned* lds, int tid) {
#pragma unroll
    for (int i = 0; i < N; ++i) {
      int p = tid + i * 256;
      int n = p % COLS, d = p / COLS;
      lds[n * LDS_PITCH + d] = bf16pair(lo[i], hi[i]);
    }
  }
};

// Async DMA of a 64x32 bf16 tile (row-major dwords) straight into LDS:
// one global_load_async_to_lds_b128 per thread, no VGPR round-trip (ASYNCcnt).
__device__ __forceinline__ void stage_async_bf16(unsigned* lds, const unsigned* __restrict__ src,
                                                 long src_stride_dw, int tid) {
  int row = tid >> 2, q = (tid & 3) * 4;                 // 4 dwords = 16B per lane
  unsigned lds_off = (unsigned)(unsigned long long)(lds + row * LDS_PITCH + q);
  const unsigned* g = src + (long)row * src_stride_dw + q;
  asm volatile("global_load_async_to_lds_b128 %0, %1, off"
               :: "v"(lds_off), "v"((unsigned long long)g)
               : "memory");
}
__device__ __forceinline__ void wait_async0() {
  asm volatile("s_wait_asynccnt 0x0" ::: "memory");
}

// A fragment (16x32 bf16): lane<16 -> M=lane, K {0..7,16..23}; lane>=16 -> K {8..15,24..31}
__device__ __forceinline__ v16bf load_a_frag(const unsigned* __restrict__ t, int lane) {
  const unsigned* rp = t + (lane & 15) * LDS_PITCH;
  int kh4 = (lane >> 4) << 2;
  v8u u;
#pragma unroll
  for (int v = 0; v < 8; ++v) u[v] = rp[(v & 3) + ((v >> 2) << 3) + kh4];
  return __builtin_bit_cast(v16bf, u);
}

// B fragment (32x16 bf16): lane<16 -> N=lane, K {0..7,16..23}; lane>=16 -> K {8..15,24..31}
__device__ __forceinline__ v16bf load_b_frag(const unsigned* __restrict__ t, int lane) {
  const unsigned* rp = t + (lane & 15) * LDS_PITCH + ((lane >> 4) << 2);
  v8u u;
#pragma unroll
  for (int v = 0; v < 8; ++v) u[v] = rp[(v & 3) + ((v >> 2) << 3)];
  return __builtin_bit_cast(v16bf, u);
}

// One K-step for a wave: wave (wm, wn) owns rows [wm*16,+16), cols [wn*64,+64)
__device__ __forceinline__ void mma_step(const unsigned* __restrict__ aT,
                                         const unsigned* __restrict__ bT,
                                         int wm, int wn, int lane, v8f acc[4]) {
  v16bf a = load_a_frag(aT + wm * 16 * LDS_PITCH, lane);
#pragma unroll
  for (int j = 0; j < 4; ++j) {
    v16bf b = load_b_frag(bT + (wn * 64 + j * 16) * LDS_PITCH, lane);
    acc[j] = __builtin_amdgcn_wmma_f32_16x16x32_bf16(false, a, false, b,
                                                     (short)0, acc[j], false, false);
  }
}

// C/D 16x16 f32 tile: VGPR r, lane: M = r + 8*(lane>>4), N = lane&15
__device__ __forceinline__ void store_tile_f32(float* __restrict__ C, long ldc, int lane,
                                               const v8f& a) {
  int col = lane & 15, rb = (lane >> 4) * 8;
#pragma unroll
  for (int r = 0; r < 8; ++r) C[(long)(rb + r) * ldc + col] = a[r];
}
__device__ __forceinline__ void store_tile_bf16(unsigned short* __restrict__ C, long ldc, int lane,
                                                const v8f& a) {
  int col = lane & 15, rb = (lane >> 4) * 8;
#pragma unroll
  for (int r = 0; r < 8; ++r) C[(long)(rb + r) * ldc + col] = bf16one(a[r]);
}

// ================================================================ kernel 1
// scores[b] = dec[b] @ enc[b]^T  -> raw logits into attn region of d_out
__global__ __launch_bounds__(256) void scores_kernel(const float* __restrict__ dec,
                                                     const float* __restrict__ enc,
                                                     float* __restrict__ attn) {
  __shared__ alignas(16) unsigned aT[2][64 * LDS_PITCH];
  __shared__ alignas(16) unsigned bT[2][128 * LDS_PITCH];
  const int tid = threadIdx.x, lane = tid & 31, wave = tid >> 5;
  const int wm = wave >> 1, wn = wave & 1;
  const int b = blockIdx.z;
  const long m0 = (long)blockIdx.y * 64;
  const long n0 = (long)blockIdx.x * 128;
  const float* A  = dec + (long)b * S_LEN * HDIM + m0 * HDIM;
  const float* Bm = enc + (long)b * S_LEN * HDIM + n0 * HDIM;   // B[k][n] = E[n][k]

  v8f z = {};
  v8f acc[4] = {z, z, z, z};
  RowPF<64> apf;
  RowPF<128> bpf;
  apf.load(A, HDIM, tid);
  bpf.load(Bm, HDIM, tid);
  apf.store(aT[0], tid);
  bpf.store(bT[0], tid);
  __syncthreads();

  constexpr int NS = HDIM / 32;
#pragma unroll 2
  for (int s = 0; s < NS; ++s) {
    const int cur = s & 1;
    if (s + 1 < NS) {
      apf.load(A + (s + 1) * 32, HDIM, tid);
      bpf.load(Bm + (s + 1) * 32, HDIM, tid);
    }
    mma_step(aT[cur], bT[cur], wm, wn, lane, acc);
    if (s + 1 < NS) {
      apf.store(aT[cur ^ 1], tid);
      bpf.store(bT[cur ^ 1], tid);
      __syncthreads();
    }
  }
  float* C = attn + (long)b * S_LEN * S_LEN + (m0 + wm * 16) * S_LEN + n0 + wn * 64;
#pragma unroll
  for (int j = 0; j < 4; ++j) store_tile_f32(C + j * 16, S_LEN, lane, acc[j]);
}

// ================================================================ kernel 2
// In-place row softmax over S_in=1024: one wave per row, 32 elems/lane.
__global__ __launch_bounds__(256) void softmax_kernel(float* __restrict__ attn) {
  const int lane = threadIdx.x & 31, wave = threadIdx.x >> 5;
  float* p = attn + ((long)blockIdx.x * 8 + wave) * S_LEN;
  float x[32];
  float m = -__builtin_inff();
#pragma unroll
  for (int i = 0; i < 32; ++i) { x[i] = p[lane + i * 32]; m = fmaxf(m, x[i]); }
#pragma unroll
  for (int off = 16; off > 0; off >>= 1) m = fmaxf(m, __shfl_xor(m, off, 32));
  float s = 0.f;
#pragma unroll
  for (int i = 0; i < 32; ++i) { x[i] = __expf(x[i] - m); s += x[i]; }
#pragma unroll
  for (int off = 16; off > 0; off >>= 1) s += __shfl_xor(s, off, 32);
  float inv = 1.0f / s;
#pragma unroll
  for (int i = 0; i < 32; ++i) p[lane + i * 32] = x[i] * inv;
}

// ================================================================ kernel 3
// ctx[b] = attn[b] @ enc[b]  -> bf16 scratch in d_ws
__global__ __launch_bounds__(256) void context_kernel(const float* __restrict__ attn,
                                                      const float* __restrict__ enc,
                                                      unsigned short* __restrict__ ctx) {
  __shared__ alignas(16) unsigned aT[2][64 * LDS_PITCH];
  __shared__ alignas(16) unsigned bT[2][128 * LDS_PITCH];
  const int tid = threadIdx.x, lane = tid & 31, wave = tid >> 5;
  const int wm = wave >> 1, wn = wave & 1;
  const int b = blockIdx.z;
  const long m0 = (long)blockIdx.y * 64;
  const long n0 = (long)blockIdx.x * 128;
  const float* A  = attn + (long)b * S_LEN * S_LEN + m0 * S_LEN;
  const float* Bm = enc + (long)b * S_LEN * HDIM + n0;          // B[k][n] = E[k][n]

  v8f z = {};
  v8f acc[4] = {z, z, z, z};
  RowPF<64> apf;
  KnPF<128> bpf;
  apf.load(A, S_LEN, tid);
  bpf.load(Bm, HDIM, tid);
  apf.store(aT[0], tid);
  bpf.store(bT[0], tid);
  __syncthreads();

  constexpr int NS = S_LEN / 32;
#pragma unroll 2
  for (int s = 0; s < NS; ++s) {
    const int cur = s & 1;
    if (s + 1 < NS) {
      apf.load(A + (s + 1) * 32, S_LEN, tid);
      bpf.load(Bm + (long)(s + 1) * 32 * HDIM, HDIM, tid);
    }
    mma_step(aT[cur], bT[cur], wm, wn, lane, acc);
    if (s + 1 < NS) {
      apf.store(aT[cur ^ 1], tid);
      bpf.store(bT[cur ^ 1], tid);
      __syncthreads();
    }
  }
  unsigned short* C = ctx + ((long)b * S_LEN + m0 + wm * 16) * HDIM + n0 + wn * 64;
#pragma unroll
  for (int j = 0; j < 4; ++j) store_tile_bf16(C + j * 16, HDIM, lane, acc[j]);
}

// ================================================================ kernel 4
// out = tanh([ctx | dec] @ W + b), flattened M = B*S_OUT = 65536.
// ctx half of A is already bf16 -> async DMA straight into LDS.
__global__ __launch_bounds__(256) void linear_kernel(const unsigned* __restrict__ ctx,
                                                     const float* __restrict__ dec,
                                                     const float* __restrict__ W,
                                                     const float* __restrict__ bias,
                                                     float* __restrict__ out) {
  __shared__ alignas(16) unsigned aT[2][64 * LDS_PITCH];
  __shared__ alignas(16) unsigned bT[2][128 * LDS_PITCH];
  const int tid = threadIdx.x, lane = tid & 31, wave = tid >> 5;
  const int wm = wave >> 1, wn = wave & 1;
  const long m0 = (long)blockIdx.y * 64;   // flat (b, s_out) row
  const long n0 = (long)blockIdx.x * 128;

  v8f z = {};
  v8f acc[4] = {z, z, z, z};
  RowPF<64> apf;
  KnPF<128> bpf;
  stage_async_bf16(aT[0], ctx + m0 * (HDIM / 2), HDIM / 2, tid);  // k0 = 0: ctx half
  bpf.load(W + n0, HDIM, tid);
  bpf.store(bT[0], tid);
  wait_async0();
  __syncthreads();

  constexpr int NS = (2 * HDIM) / 32;
#pragma unroll 2
  for (int s = 0; s < NS; ++s) {
    const int cur = s & 1;
    const int k0n = (s + 1) * 32;
    bool use_async = false;
    if (s + 1 < NS) {
      if (k0n < HDIM) {
        stage_async_bf16(aT[cur ^ 1], ctx + m0 * (HDIM / 2) + k0n / 2, HDIM / 2, tid);
        use_async = true;
      } else {
        apf.load(dec + m0 * HDIM + (k0n - HDIM), HDIM, tid);
      }
      bpf.load(W + (long)k0n * HDIM + n0, HDIM, tid);
    }
    mma_step(aT[cur], bT[cur], wm, wn, lane, acc);
    if (s + 1 < NS) {
      if (!use_async) apf.store(aT[cur ^ 1], tid);
      bpf.store(bT[cur ^ 1], tid);
      wait_async0();
      __syncthreads();
    }
  }
  const int col = lane & 15, rb = (lane >> 4) * 8;
  float* C = out + (m0 + wm * 16) * HDIM + n0 + wn * 64;
#pragma unroll
  for (int j = 0; j < 4; ++j) {
    float bb = bias[n0 + wn * 64 + j * 16 + col];
#pragma unroll
    for (int r = 0; r < 8; ++r)
      C[(long)(rb + r) * HDIM + j * 16 + col] = tanhf(acc[j][r] + bb);
  }
}

// ================================================================ launch
extern "C" void kernel_launch(void* const* d_in, const int* in_sizes, int n_in,
                              void* d_out, int out_size, void* d_ws, size_t ws_size,
                              hipStream_t stream) {
  const float* enc  = (const float*)d_in[0];  // [64,1024,512]
  const float* dec  = (const float*)d_in[1];  // [64,1024,512]
  const float* W    = (const float*)d_in[2];  // [1024,512]
  const float* bias = (const float*)d_in[3];  // [512]

  float* out  = (float*)d_out;                          // [64,1024,512]
  float* attn = out + (long)NB * S_LEN * HDIM;          // [64,1024,1024]

  // bf16 context scratch: 64*1024*512*2B = 64 MB (assumes ws_size >= 64 MB)
  unsigned short* ctx = (unsigned short*)d_ws;

  scores_kernel<<<dim3(S_LEN / 128, S_LEN / 64, NB), 256, 0, stream>>>(dec, enc, attn);
  softmax_kernel<<<(NB * S_LEN) / 8, 256, 0, stream>>>(attn);
  context_kernel<<<dim3(HDIM / 128, S_LEN / 64, NB), 256, 0, stream>>>(attn, enc, ctx);
  linear_kernel<<<dim3(HDIM / 128, (NB * S_LEN) / 64), 256, 0, stream>>>(
      (const unsigned*)d_ws, dec, W, bias, out);
}
